// WindowMultiHeadSelfAttention_6605659701801
// MI455X (gfx1250) — matrix operand order repaired and checked by reading, not compiled, not run on hardware
//
#include <hip/hip_runtime.h>

#define DIM   256
#define NH    8
#define HD    32
#define NT    64
#define NWIN  512
#define SCALE 0.17677669529663687f  // HD^-0.5

typedef __attribute__((ext_vector_type(16))) __bf16        v16bf;
typedef __attribute__((ext_vector_type(8)))  float         v8f;
typedef __attribute__((ext_vector_type(16))) unsigned short v16u;
typedef __attribute__((ext_vector_type(8)))  unsigned short v8u;

#define WMMA_BF16(a, b, c) \
  __builtin_amdgcn_wmma_f32_16x16x32_bf16(false, (a), false, (b), (short)0, (c), false, false)

// native hardware f32 -> bf16 convert (RTNE); backend emits v_cvt_pk_bf16_f32
__device__ __forceinline__ unsigned short f2bf(float f) {
  __bf16 h = (__bf16)f;
  return __builtin_bit_cast(unsigned short, h);
}

// async DMA: 16 bytes global -> LDS, tracked by ASYNCcnt (CDNA5 GV mode, saddr=off)
__device__ __forceinline__ void async_g2l_b128(unsigned lds_byte_off, const void* g) {
  unsigned long long ga = (unsigned long long)(uintptr_t)g;
  asm volatile("global_load_async_to_lds_b128 %0, %1, off"
               :: "v"(lds_byte_off), "v"(ga) : "memory");
}
__device__ __forceinline__ void wait_asynccnt0() {
  asm volatile("s_wait_asynccnt 0x0" ::: "memory");
}

// A fragment (16x32 bf16), row-major source, tile at (row0, k0).
// ISA layout: lane%16 = row M; lanes<16 hold K = k0+{0..7, 16..23}, lanes>=16 K = k0+{8..15, 24..31}.
__device__ __forceinline__ v16bf load_frag_a(const unsigned short* s, int ld,
                                             int row0, int k0, int lane) {
  const unsigned short* p = s + (row0 + (lane & 15)) * ld + k0 + ((lane >> 4) << 3);
  v8u lo = *(const v8u*)p;
  v8u hi = *(const v8u*)(p + 16);
  v16u t;
#pragma unroll
  for (int i = 0; i < 8; ++i) { t[i] = lo[i]; t[8 + i] = hi[i]; }
  return __builtin_bit_cast(v16bf, t);
}

// B fragment (32x16 bf16) where B[k][n] = S[n0+n][k0+k], S row-major (stride ld).
// ISA layout: lane%16 = col N; lanes<16 hold K = k0+{0..7, 16..23}, lanes>=16 K = k0+{8..15, 24..31}.
__device__ __forceinline__ v16bf load_frag_b(const unsigned short* s, int ld,
                                             int n0, int k0, int lane) {
  const unsigned short* p = s + (n0 + (lane & 15)) * ld + k0 + ((lane >> 4) << 3);
  v8u lo = *(const v8u*)p;
  v8u hi = *(const v8u*)(p + 16);
  v16u t;
#pragma unroll
  for (int i = 0; i < 8; ++i) { t[i] = lo[i]; t[8 + i] = hi[i]; }
  return __builtin_bit_cast(v16bf, t);
}

__global__ void cvt_weights_kernel(const float* __restrict__ wq, const float* __restrict__ wp,
                                   unsigned short* __restrict__ oq, unsigned short* __restrict__ op_) {
  int i = blockIdx.x * blockDim.x + threadIdx.x;
  if (i < 3 * DIM * DIM) oq[i] = f2bf(wq[i]);
  if (i < DIM * DIM)     op_[i] = f2bf(wp[i]);
}

__global__ __launch_bounds__(256)
void wmsa_kernel(const float* __restrict__ x, const float* __restrict__ mask,
                 const unsigned short* __restrict__ wqkv_bf, const float* __restrict__ b_qkv,
                 const unsigned short* __restrict__ wproj_bf, const float* __restrict__ b_proj,
                 float* __restrict__ out) {
  extern __shared__ unsigned short smem[];
  unsigned short* xs   = smem;                    // [64][256] bf16 x        (reused as ob)
  unsigned short* qb   = smem + 1 * NT * DIM;     // [64][256] Q (pre-scaled)
  unsigned short* kb   = smem + 2 * NT * DIM;     // [64][256] K
  unsigned short* vt   = smem + 3 * NT * DIM;     // [256][64] V transposed
  unsigned short* attn = smem + 4 * NT * DIM;     // [8][16][64] per-head row-block scratch
  float* ml            = (float*)(smem + 4 * NT * DIM + NH * 16 * NT);  // [64][64] mask f32
  unsigned short* ob   = smem;                    // [64][256] head-concat attn@V (aliases xs)

  const int tid  = threadIdx.x;
  const int lane = tid & 31;
  const int wave = tid >> 5;
  const int widx = blockIdx.x;

  const float* xg = x    + (size_t)widx * NT * DIM;
  const float* mg = mask + (size_t)(widx % NWIN) * NT * NT;
  float*       og = out  + (size_t)widx * NT * DIM;

  // ---- phase 0: kick off async DMA of the mask tile into LDS -----------------
  // 16 KB, 1024 x b128; overlaps with phases 1-2, consumed by the softmax.
  {
    unsigned mbase = (unsigned)(uintptr_t)ml;     // low 32 bits = LDS byte address
#pragma unroll
    for (int i = 0; i < 4; ++i) {
      int idx = tid + i * 256;                    // 16-byte chunk index
      async_g2l_b128(mbase + idx * 16, (const char*)mg + idx * 16);
    }
  }

  // ---- phase 1: load x, convert to bf16 in LDS -------------------------------
  {
    const float4* xv = (const float4*)xg;
#pragma unroll
    for (int i = 0; i < 16; ++i) {
      int idx = tid + i * 256;              // 4096 float4 total
      float4 f = xv[idx];
      ushort4 h;
      h.x = f2bf(f.x); h.y = f2bf(f.y); h.z = f2bf(f.z); h.w = f2bf(f.w);
      *(ushort4*)(xs + idx * 4) = h;
    }
  }
  __syncthreads();

  // ---- phase 2: QKV GEMM  qkv = x @ Wqkv^T + b -------------------------------
  // output [64 tokens][768 ch]: 4 row-tiles x 48 col-tiles, K = 256 (8 WMMA steps)
  // each wave has a fixed row-tile (t & 3 invariant) -> A fragments stay hoisted in VGPRs
  for (int t = wave; t < 192; t += 8) {
    int rt = t & 3, ct = t >> 2;
    int row0 = rt * 16, oc0 = ct * 16;
    v8f acc = {};
#pragma unroll
    for (int kk = 0; kk < 8; ++kk) {
      v16bf a = load_frag_a(xs, DIM, row0, kk * 32, lane);
      v16bf b = load_frag_b(wqkv_bf, DIM, oc0, kk * 32, lane);
      acc = WMMA_BF16(a, b, acc);
    }
    int col  = oc0 + (lane & 15);
    float bi = b_qkv[col];
    int r0   = row0 + ((lane >> 4) << 3);
    if (oc0 < 256) {                     // Q, fold in softmax scale
#pragma unroll
      for (int j = 0; j < 8; ++j) qb[(r0 + j) * DIM + col] = f2bf((acc[j] + bi) * SCALE);
    } else if (oc0 < 512) {              // K
      int c = col - 256;
#pragma unroll
      for (int j = 0; j < 8; ++j) kb[(r0 + j) * DIM + c] = f2bf(acc[j] + bi);
    } else {                             // V stored transposed: vt[channel][token]
      int c = col - 512;
#pragma unroll
      for (int j = 0; j < 8; ++j) vt[c * NT + (r0 + j)] = f2bf(acc[j] + bi);
    }
  }
  wait_asynccnt0();        // this wave's mask DMA chunks are in LDS
  __syncthreads();         // ... and every other wave's too

  // ---- phase 3: per-head attention (wave h == head h) ------------------------
  {
    const int h = wave;
    unsigned short* at = attn + h * 16 * NT;   // 16x64 row-block scratch for this head
    int cbase = lane & 15;

    v16bf bk[4];                               // B = K^T (key tokens x channel-K)
#pragma unroll
    for (int ct = 0; ct < 4; ++ct)
      bk[ct] = load_frag_b(kb, DIM, ct * 16, h * HD, lane);

    v16bf bv[2][2];                            // B = V (key tokens K-dim x hd cols), from vt
#pragma unroll
    for (int nt = 0; nt < 2; ++nt) {
      bv[nt][0] = load_frag_b(vt, NT, h * HD + nt * 16, 0, lane);
      bv[nt][1] = load_frag_b(vt, NT, h * HD + nt * 16, 32, lane);
    }

    for (int rt = 0; rt < 4; ++rt) {
      v16bf aq = load_frag_a(qb, DIM, rt * 16, h * HD, lane);
      v8f sc[4];
#pragma unroll
      for (int ct = 0; ct < 4; ++ct) {
        v8f z = {};
        sc[ct] = WMMA_BF16(aq, bk[ct], z);
      }
      int r0  = rt * 16 + ((lane >> 4) << 3);      // absolute row
      int lr0 = ((lane >> 4) << 3);                // row within scratch block
#pragma unroll
      for (int ct = 0; ct < 4; ++ct)
#pragma unroll
        for (int j = 0; j < 8; ++j)
          sc[ct][j] += ml[(r0 + j) * NT + ct * 16 + cbase];

      // softmax: each row lives across one 16-lane group (4 col-tiles wide)
#pragma unroll
      for (int j = 0; j < 8; ++j) {
        float m = fmaxf(fmaxf(sc[0][j], sc[1][j]), fmaxf(sc[2][j], sc[3][j]));
        for (int s = 1; s < 16; s <<= 1) m = fmaxf(m, __shfl_xor(m, s, 16));
        float e0 = __expf(sc[0][j] - m), e1 = __expf(sc[1][j] - m);
        float e2 = __expf(sc[2][j] - m), e3 = __expf(sc[3][j] - m);
        float su = e0 + e1 + e2 + e3;
        for (int s = 1; s < 16; s <<= 1) su += __shfl_xor(su, s, 16);
        float ri = 1.0f / su;
        sc[0][j] = e0 * ri; sc[1][j] = e1 * ri; sc[2][j] = e2 * ri; sc[3][j] = e3 * ri;
      }
#pragma unroll
      for (int ct = 0; ct < 4; ++ct)
#pragma unroll
        for (int j = 0; j < 8; ++j)
          at[(lr0 + j) * NT + ct * 16 + cbase] = f2bf(sc[ct][j]);

      // attn @ V for this row-block: K = 64 key tokens (2 steps), N = hd (2 tiles)
      v16bf a0 = load_frag_a(at, NT, 0, 0, lane);
      v16bf a1 = load_frag_a(at, NT, 0, 32, lane);
#pragma unroll
      for (int nt = 0; nt < 2; ++nt) {
        v8f o = {};
        o = WMMA_BF16(a0, bv[nt][0], o);
        o = WMMA_BF16(a1, bv[nt][1], o);
        int col = h * HD + nt * 16 + cbase;
#pragma unroll
        for (int j = 0; j < 8; ++j) ob[(r0 + j) * DIM + col] = f2bf(o[j]);
      }
    }
  }
  __syncthreads();

  // ---- phase 4: output projection  out = ob @ Wproj^T + b --------------------
  for (int t = wave; t < 64; t += 8) {
    int rt = t & 3, ct = t >> 2;
    int row0 = rt * 16, col0 = ct * 16;
    v8f acc = {};
#pragma unroll
    for (int kk = 0; kk < 8; ++kk) {
      v16bf a = load_frag_a(ob, DIM, row0, kk * 32, lane);
      v16bf b = load_frag_b(wproj_bf, DIM, col0, kk * 32, lane);
      acc = WMMA_BF16(a, b, acc);
    }
    int col  = col0 + (lane & 15);
    float bi = b_proj[col];
    int r0   = row0 + ((lane >> 4) << 3);
#pragma unroll
    for (int j = 0; j < 8; ++j) og[(r0 + j) * DIM + col] = acc[j] + bi;
  }
}

extern "C" void kernel_launch(void* const* d_in, const int* in_sizes, int n_in,
                              void* d_out, int out_size, void* d_ws, size_t ws_size,
                              hipStream_t stream) {
  const float* x     = (const float*)d_in[0];
  const float* mask  = (const float*)d_in[1];
  const float* wqkv  = (const float*)d_in[2];
  const float* bqkv  = (const float*)d_in[3];
  const float* wproj = (const float*)d_in[4];
  const float* bproj = (const float*)d_in[5];
  float* out = (float*)d_out;

  unsigned short* wq_bf = (unsigned short*)d_ws;        // 768*256 bf16
  unsigned short* wp_bf = wq_bf + 3 * DIM * DIM;        // 256*256 bf16

  cvt_weights_kernel<<<(3 * DIM * DIM + 255) / 256, 256, 0, stream>>>(wqkv, wproj, wq_bf, wp_bf);

  // 4 x 32KB (x/ob, Q, K, Vt) + 16KB attn scratch + 16KB mask = 160 KB
  //   -> exactly 2 blocks per 320KB WGP
  size_t lds_bytes = (size_t)(4 * NT * DIM + NH * 16 * NT) * sizeof(unsigned short)
                   + (size_t)(NT * NT) * sizeof(float);
  hipFuncSetAttribute((const void*)wmsa_kernel,
                      hipFuncAttributeMaxDynamicSharedMemorySize, (int)lds_bytes);
  wmsa_kernel<<<2048, 256, lds_bytes, stream>>>(x, mask, wq_bf, bqkv, wp_bf, bproj, out);
}
